// SparseMoEFFN_23003844837839
// MI455X (gfx1250) — matrix operand order repaired and checked by reading, not compiled, not run on hardware
//
#include <hip/hip_runtime.h>
#include <hip/hip_bf16.h>

typedef __attribute__((ext_vector_type(16))) __bf16 v16bf;
typedef __attribute__((ext_vector_type(8)))  float  v8f;
typedef unsigned short u16;

#define NTOK   4096
#define DM     512
#define DFF    2048
#define NE     8
#define TILE_M 32         // tokens per block (2 WMMA M-tiles)
#define FFC    128        // ff chunk width = 8 waves * 16
#define MAXT   (NTOK / TILE_M)   // 128 tiles per expert max

union Frag {
    v16bf v;
    uint4 q[2];
};

__device__ __forceinline__ u16 f2bf(float f) {
    union { float f; unsigned u; } cv;
    cv.f = f;
    unsigned u = cv.u;
    unsigned r = u + 0x7FFFu + ((u >> 16) & 1u);   // RNE
    return (u16)(r >> 16);
}

// ---------------- conversion kernels ----------------

__global__ void convert_x_kernel(const float* __restrict__ src, u16* __restrict__ dst, int n) {
    int i = blockIdx.x * blockDim.x + threadIdx.x;
    if (i < n) dst[i] = f2bf(src[i]);
}

// Swizzle fp32 weights [E][K][N] into bf16 B-fragment tiles:
// per expert, tile (kt,nt) covers K=[kt*32,+32), N=[nt*16,+16).
// dst[((kt*ntiles+nt)*32 + lane)*16 + j] = src[k=kt*32+8*(lane/16)+(j%8)+16*(j/8)][n=nt*16+lane%16]
__global__ void convert_w_kernel(const float* __restrict__ src, u16* __restrict__ dst,
                                 int K, int N) {
    size_t idx = (size_t)blockIdx.x * blockDim.x + threadIdx.x;
    size_t per = (size_t)K * N;
    if (idx >= (size_t)NE * per) return;
    int    e    = (int)(idx / per);
    size_t r    = idx % per;
    int    j    = (int)(r & 15);
    int    lane = (int)((r >> 4) & 31);
    size_t tile = r >> 9;
    int ntiles  = N >> 4;
    int nt = (int)(tile % ntiles);
    int kt = (int)(tile / ntiles);
    int k = kt * 32 + ((lane >> 4) << 3) + (j & 7) + ((j >> 3) << 4);
    int n = nt * 16 + (lane & 15);
    dst[idx] = f2bf(src[(size_t)e * per + (size_t)k * N + n]);
}

// ---------------- router ----------------

__global__ void router_kernel(const float* __restrict__ x, const float* __restrict__ rw,
                              const float* __restrict__ rb,
                              int* __restrict__ counts, int* __restrict__ toks,
                              float* __restrict__ wgts) {
    int n = blockIdx.x * blockDim.x + threadIdx.x;
    if (n >= NTOK) return;
    float acc[NE];
#pragma unroll
    for (int e = 0; e < NE; ++e) acc[e] = rb[e];
    const float* xr = x + (size_t)n * DM;
    for (int k = 0; k < DM; ++k) {
        float xv = xr[k];
#pragma unroll
        for (int e = 0; e < NE; ++e) acc[e] = fmaf(xv, rw[k * NE + e], acc[e]);
    }
    float mx = acc[0];
#pragma unroll
    for (int e = 1; e < NE; ++e) mx = fmaxf(mx, acc[e]);
    float p[NE];
    float z = 0.f;
#pragma unroll
    for (int e = 0; e < NE; ++e) { p[e] = expf(acc[e] - mx); z += p[e]; }
    float inv = 1.f / z;
#pragma unroll
    for (int e = 0; e < NE; ++e) p[e] *= inv;
    // top-2, lowest index wins ties (matches lax.top_k)
    int i0 = 0; float p0 = p[0];
#pragma unroll
    for (int e = 1; e < NE; ++e) if (p[e] > p0) { p0 = p[e]; i0 = e; }
    p[i0] = -1.f;
    int i1 = 0; float p1 = p[0];
#pragma unroll
    for (int e = 1; e < NE; ++e) if (p[e] > p1) { p1 = p[e]; i1 = e; }
    float s = p0 + p1 + 1e-9f;
    float w0 = p0 / s, w1 = p1 / s;
    int pos0 = atomicAdd(&counts[i0], 1);
    toks[i0 * NTOK + pos0] = n;
    wgts[i0 * NTOK + pos0] = w0;
    int pos1 = atomicAdd(&counts[i1], 1);
    toks[i1 * NTOK + pos1] = n;
    wgts[i1 * NTOK + pos1] = w1;
}

// ---------------- MoE FFN (bf16 WMMA, M=32) ----------------

__global__ __launch_bounds__(256) void moe_ffn_kernel(
    const u16* __restrict__ xbf,   // [NTOK][DM] bf16 row-major
    const u16* __restrict__ w1s,   // [E] swizzled [K=DM][N=DFF]
    const u16* __restrict__ w2s,   // [E] swizzled [K=DFF][N=DM]
    const float* __restrict__ b1,  // [E][DFF]
    const float* __restrict__ b2,  // [E][DM]
    const int* __restrict__ counts,
    const int* __restrict__ toks,
    const float* __restrict__ wgts,
    float* __restrict__ out)       // [NTOK][DM] fp32, pre-zeroed
{
    __shared__ u16   xtile[TILE_M][DM];   // 32 KB: gathered activations (bf16)
    __shared__ u16   hbuf[TILE_M][FFC];   //  8 KB: gelu(h) chunk (bf16)
    __shared__ int   stok[TILE_M];
    __shared__ float swgt[TILE_M];

    const int e = blockIdx.x / MAXT;
    const int t = blockIdx.x % MAXT;
    const int cnt = counts[e];
    if (t * TILE_M >= cnt) return;

    const int tid  = threadIdx.x;
    const int lane = tid & 31;
    const int w    = tid >> 5;      // wave id 0..7
    const int m16  = lane & 15;     // N-col (B/C) or M-row (A)
    const int hi   = lane >> 4;     // K-half selector

    if (tid < TILE_M) {
        int idx = t * TILE_M + tid;
        int tok = 0; float wt = 0.f;
        if (idx < cnt) { tok = toks[e * NTOK + idx]; wt = wgts[e * NTOK + idx]; }
        stok[tid] = tok;
        swgt[tid] = wt;
    }
    __syncthreads();

    // gather 32 token rows -> LDS (coalesced b128): 8 threads/row, 8 uint4 each
    {
        int row = tid >> 3;     // 0..31
        int g   = tid & 7;      // 0..7
        const uint4* src = (const uint4*)(xbf + (size_t)stok[row] * DM);
        uint4* dst = (uint4*)(&xtile[row][0]);
#pragma unroll
        for (int i = 0; i < 8; ++i) dst[g + 8 * i] = src[g + 8 * i];
    }
    __syncthreads();

    v8f yacc0[4], yacc1[4];
#pragma unroll
    for (int i = 0; i < 4; ++i) {
        yacc0[i] = (v8f){0.f, 0.f, 0.f, 0.f, 0.f, 0.f, 0.f, 0.f};
        yacc1[i] = (v8f){0.f, 0.f, 0.f, 0.f, 0.f, 0.f, 0.f, 0.f};
    }

    const u16* w1base = w1s + (size_t)e * DM * DFF;
    const u16* w2base = w2s + (size_t)e * DFF * DM;

    for (int c = 0; c < DFF / FFC; ++c) {      // 16 ff chunks
        // ---- GEMM1: h[32,16] for cols n0 = c*FFC + w*16, K = DM ----
        v8f h0 = (v8f){0.f, 0.f, 0.f, 0.f, 0.f, 0.f, 0.f, 0.f};
        v8f h1 = (v8f){0.f, 0.f, 0.f, 0.f, 0.f, 0.f, 0.f, 0.f};
        const int nt1 = c * (FFC / 16) + w;    // ntiles(W1) = 128
        // per-chunk base pointer; kb advances by constant 8192 uint4 (128 KB)
        const uint4* bp1 = (const uint4*)(w1base + ((size_t)nt1 * 32 + lane) * 16);
#pragma unroll
        for (int kb = 0; kb < DM / 32; ++kb) {
            Frag a0, a1, b;
            b.q[0]  = bp1[kb * 8192];
            b.q[1]  = bp1[kb * 8192 + 1];
            a0.q[0] = *(const uint4*)(&xtile[m16][kb * 32 + 8 * hi]);
            a0.q[1] = *(const uint4*)(&xtile[m16][kb * 32 + 16 + 8 * hi]);
            a1.q[0] = *(const uint4*)(&xtile[16 + m16][kb * 32 + 8 * hi]);
            a1.q[1] = *(const uint4*)(&xtile[16 + m16][kb * 32 + 16 + 8 * hi]);
            h0 = __builtin_amdgcn_wmma_f32_16x16x32_bf16(
                false, a0.v, false, b.v, (short)0, h0, false, false);
            h1 = __builtin_amdgcn_wmma_f32_16x16x32_bf16(
                false, a1.v, false, b.v, (short)0, h1, false, false);
        }
        // bias + exact GELU + bf16 store to LDS (both M-halves)
        float b1v = b1[e * DFF + c * FFC + w * 16 + m16];
#pragma unroll
        for (int v = 0; v < 8; ++v) {
            float hv = h0[v] + b1v;
            hv = 0.5f * hv * (1.0f + erff(hv * 0.70710678118654752f));
            hbuf[v + 8 * hi][w * 16 + m16] = f2bf(hv);
            float gv = h1[v] + b1v;
            gv = 0.5f * gv * (1.0f + erff(gv * 0.70710678118654752f));
            hbuf[16 + v + 8 * hi][w * 16 + m16] = f2bf(gv);
        }
        __syncthreads();

        // ---- GEMM2: y[32, 64 cols owned by wave] += h_chunk @ W2[chunk,:] ----
        // base: kt0 = c*4 ; element offset (kt*1024 + nt2*32 + lane)*16
        const uint4* bp2 = (const uint4*)(w2base + ((size_t)(c * 4) * 16384 + (size_t)lane * 16));
#pragma unroll
        for (int kk = 0; kk < FFC / 32; ++kk) {
            Frag a0, a1;
            a0.q[0] = *(const uint4*)(&hbuf[m16][kk * 32 + 8 * hi]);
            a0.q[1] = *(const uint4*)(&hbuf[m16][kk * 32 + 16 + 8 * hi]);
            a1.q[0] = *(const uint4*)(&hbuf[16 + m16][kk * 32 + 8 * hi]);
            a1.q[1] = *(const uint4*)(&hbuf[16 + m16][kk * 32 + 16 + 8 * hi]);
#pragma unroll
            for (int tt = 0; tt < 4; ++tt) {
                int nt2 = w * 4 + tt;          // ntiles(W2) = 32
                Frag b;
                // uint4 offset: kk*1024 (16 KB elems) + nt2*32
                b.q[0] = bp2[kk * 1024 + nt2 * 32];
                b.q[1] = bp2[kk * 1024 + nt2 * 32 + 1];
                yacc0[tt] = __builtin_amdgcn_wmma_f32_16x16x32_bf16(
                    false, a0.v, false, b.v, (short)0, yacc0[tt], false, false);
                yacc1[tt] = __builtin_amdgcn_wmma_f32_16x16x32_bf16(
                    false, a1.v, false, b.v, (short)0, yacc1[tt], false, false);
            }
        }
        __syncthreads();
    }

    // ---- epilogue: +b2, scale by routing weight, scatter-add ----
#pragma unroll
    for (int tt = 0; tt < 4; ++tt) {
        int n0 = w * 64 + tt * 16 + m16;
        float b2v = b2[e * DM + n0];
#pragma unroll
        for (int v = 0; v < 8; ++v) {
            int m = v + 8 * hi;
            float val0 = (yacc0[tt][v] + b2v) * swgt[m];
            atomicAdd(&out[(size_t)stok[m] * DM + n0], val0);
            float val1 = (yacc1[tt][v] + b2v) * swgt[16 + m];
            atomicAdd(&out[(size_t)stok[16 + m] * DM + n0], val1);
        }
    }
}

// ---------------- host launcher ----------------

extern "C" void kernel_launch(void* const* d_in, const int* in_sizes, int n_in,
                              void* d_out, int out_size, void* d_ws, size_t ws_size,
                              hipStream_t stream) {
    const float* x   = (const float*)d_in[0];  // [2,2048,512]
    const float* rw  = (const float*)d_in[1];  // [512,8]
    const float* rb  = (const float*)d_in[2];  // [8]
    const float* W1  = (const float*)d_in[3];  // [8,512,2048]
    const float* b1  = (const float*)d_in[4];  // [8,2048]
    const float* W2  = (const float*)d_in[5];  // [8,2048,512]
    const float* b2  = (const float*)d_in[6];  // [8,512]
    float* out = (float*)d_out;

    // workspace layout
    char* ws = (char*)d_ws;
    int*   counts = (int*)ws;                               // 32 B (pad to 256)
    int*   toks   = (int*)(ws + 256);                       // 128 KB
    float* wgts   = (float*)(ws + 256 + (NE * NTOK * 4));   // 128 KB
    u16*   xbf    = (u16*)(ws + 256 + 2 * (NE * NTOK * 4)); // 4 MB
    u16*   w1s    = xbf + (size_t)NTOK * DM;                // 16 MB
    u16*   w2s    = w1s + (size_t)NE * DM * DFF;            // 16 MB

    hipMemsetAsync(counts, 0, NE * sizeof(int), stream);
    hipMemsetAsync(out, 0, (size_t)out_size * sizeof(float), stream);

    // 1) x -> bf16
    {
        int n = NTOK * DM;
        convert_x_kernel<<<(n + 255) / 256, 256, 0, stream>>>(x, xbf, n);
    }
    // 2) W1, W2 -> bf16 swizzled B-fragment layout
    {
        size_t n = (size_t)NE * DM * DFF;
        convert_w_kernel<<<(unsigned)((n + 255) / 256), 256, 0, stream>>>(W1, w1s, DM, DFF);
        convert_w_kernel<<<(unsigned)((n + 255) / 256), 256, 0, stream>>>(W2, w2s, DFF, DM);
    }
    // 3) router: top-2 assignment lists
    router_kernel<<<NTOK / 256, 256, 0, stream>>>(x, rw, rb, counts, toks, wgts);

    // 4) sparse expert FFN with bf16 WMMA (32-token tiles)
    moe_ffn_kernel<<<NE * MAXT, 256, 0, stream>>>(
        xbf, w1s, w2s, b1, b2, counts, toks, wgts, out);
}
